// I2HOFI_90666759618614
// MI455X (gfx1250) — compile-verified, matrix operand order checked by reading, not run on hardware
//
#include <hip/hip_runtime.h>
#include <hip/hip_bf16.h>
#include <cstdint>

// ---------------------------------------------------------------------------
// I2HOFI pipeline for gfx1250 (MI455X, wave32, WMMA).
// Heavy GEMMs: v_wmma_f32_16x16x32_bf16, fp32 accumulation, double-buffered
// LDS staging with fragment-order B tiles (b128 LDS traffic only).
// APPNP propagation uses the closed form A=(ones+I)/(N+1).
// Inter-graph attention stages its tile with global_load_async_to_lds_b128.
// ---------------------------------------------------------------------------

typedef __attribute__((ext_vector_type(16))) __bf16        v16bf;
typedef __attribute__((ext_vector_type(8)))  float         v8f;
typedef __attribute__((ext_vector_type(4)))  float         fx4;
typedef __attribute__((ext_vector_type(4)))  unsigned int  u32x4;
typedef __attribute__((ext_vector_type(2)))  unsigned int  u32x2;

__device__ __forceinline__ __bf16 f2bf(float x) { return (__bf16)x; }

__device__ __forceinline__ v8f wmma_bf16(v16bf a, v16bf b, v8f c) {
    return __builtin_amdgcn_wmma_f32_16x16x32_bf16(false, a, false, b, (short)0, c, false, false);
}

// A fragment from a row-major LDS tile with leading dim 40 (rows 16B aligned).
// Lane layout (ISA 7.12.2): lanes 0-15 -> M=lane, K in {0..7,16..23};
// lanes 16-31 -> M=lane-16, K in {8..15,24..31}. Two contiguous 16B runs.
__device__ __forceinline__ v16bf frag_row40(const __bf16* tile, int m16base, int lane) {
    const __bf16* row = tile + (m16base + (lane & 15)) * 40;
    int k0 = ((lane >> 4) & 1) << 3;
    union { u32x4 q[2]; v16bf v; } f;
    f.q[0] = *(const u32x4*)(row + k0);
    f.q[1] = *(const u32x4*)(row + k0 + 16);
    return f.v;
}

// B fragment from a packed fragment-order tile: frags[nt][lane][16].
__device__ __forceinline__ v16bf frag_packed(const __bf16* frags, int nt, int lane) {
    const __bf16* p = frags + (nt << 9) + (lane << 4);
    union { u32x4 q[2]; v16bf v; } f;
    f.q[0] = *(const u32x4*)(p);
    f.q[1] = *(const u32x4*)(p + 8);
    return f.v;
}

// ---------------------------------------------------------------------------
#define BATCH   8
#define CH      2048
#define GDIM    256
#define NROI    13
#define CNODES  392
#define ROWS_G  (BATCH * NROI * CNODES)   // 40768
#define X2NODES (2 * NROI * CNODES)       // 10192
#define POOLROWS (BATCH * X2NODES)        // 81536

// ---------------------------------------------------------------------------
// ROI pooling: chained bilinear resizes collapse to <=16 taps on the 7x7 base.
// ---------------------------------------------------------------------------
__device__ __forceinline__ void axis_taps(int L, int off, int p, int* idx, float* wt) {
    float vc = (p + 0.5f) * (float)L / 7.0f - 0.5f;
    vc = fminf(fmaxf(vc, 0.f), (float)(L - 1));
    int i0 = (int)floorf(vc);
    float f = vc - (float)i0;
    int i1 = (i0 + 1 < L) ? i0 + 1 : L - 1;
    int   ii[2] = { i0, i1 };
    float ff[2] = { 1.f - f, f };
#pragma unroll
    for (int q = 0; q < 2; ++q) {
        float u = ((float)(off + ii[q]) + 0.5f) / 6.0f - 0.5f;
        u = fminf(fmaxf(u, 0.f), 6.f);
        int j0 = (int)floorf(u);
        float fu = u - (float)j0;
        int j1 = (j0 + 1 < 7) ? j0 + 1 : 6;
        idx[2 * q]     = j0; wt[2 * q]     = ff[q] * (1.f - fu);
        idx[2 * q + 1] = j1; wt[2 * q + 1] = ff[q] * fu;
    }
}

__global__ __launch_bounds__(256) void k_roi_nodes(const float* __restrict__ base,
                                                   float* __restrict__ nintra,
                                                   float* __restrict__ ninter) {
    int bid = blockIdx.x;
    int pix = bid % 49;
    int tmp = bid / 49;
    int r = tmp % NROI;
    int b = tmp / NROI;
    int py = pix / 7, px = pix % 7;
    int tid = threadIdx.x;

    int ry[4], rx[4];
    float wy[4], wx[4];
    int nt = 4;
    if (r < 12) {
        int x0, y0, w0, h0;
        if (r < 9)       { x0 = (r % 3) * 14; y0 = (r / 3) * 14; w0 = 14; h0 = 14; }
        else if (r == 9) { x0 = 0; y0 = 0;  w0 = 42; h0 = 21; }
        else if (r == 10){ x0 = 0; y0 = 21; w0 = 42; h0 = 21; }
        else             { x0 = 0; y0 = 0;  w0 = 42; h0 = 42; }
        axis_taps(h0, y0, py, ry, wy);
        axis_taps(w0, x0, px, rx, wx);
    } else {
        ry[0] = py; wy[0] = 1.f; rx[0] = px; wx[0] = 1.f; nt = 1;
    }

#pragma unroll
    for (int ci = 0; ci < 8; ++ci) {
        int ch = ci * 256 + tid;
        float acc = 0.f;
        for (int a = 0; a < nt; ++a)
            for (int c = 0; c < nt; ++c)
                acc += wy[a] * wx[c] *
                       base[(((size_t)b * 7 + ry[a]) * 7 + rx[c]) * CH + ch];
        int node = ci * 49 + pix;
        nintra[(((size_t)b * NROI + r) * CNODES + node) * GDIM + tid] = acc;
        ninter[(((size_t)b * CNODES + node) * NROI + r) * GDIM + tid] = acc;
    }
}

// ---------------------------------------------------------------------------
// GEMM: Y[rows x 256] = act(X @ W[256x256] + bias). Double-buffered LDS,
// 64x64 tile per 256-thread block (8 waves, 16x32 output per wave).
// ---------------------------------------------------------------------------
__global__ __launch_bounds__(256) void k_gemm256(const float* __restrict__ X,
                                                 const float* __restrict__ W,
                                                 const float* __restrict__ bias,
                                                 float* __restrict__ Y,
                                                 int rows, int act) {
    __shared__ __align__(16) __bf16 lA[2][64][40];
    __shared__ __align__(16) __bf16 lB[2][2048];
    int tid = threadIdx.x, lane = tid & 31, wid = tid >> 5;
    int wm = wid & 3, wn = wid >> 2;
    int row0 = blockIdx.x * 64, col0 = blockIdx.y * 64;

    // B writer: 8 consecutive k for one n -> one b128 LDS store in frag order.
    int bn  = tid & 63;
    int bkb = (tid >> 6) << 3;
    int boff = (((bn >> 4) << 9) | ((((bn & 15) | (((bkb >> 3) & 1) << 4)) << 4))) +
               ((bkb & 16) ? 8 : 0);

    fx4 va[2]; float wb[8];

#define LOAD_AB(ks_)                                                           \
    {   int ks = (ks_);                                                        \
        _Pragma("unroll")                                                      \
        for (int i = 0; i < 2; ++i) {                                          \
            int e = tid + i * 256;                                             \
            int rr = e >> 3, kq = (e & 7) << 2;                                \
            va[i] = *(const fx4*)&X[(size_t)(row0 + rr) * GDIM + ks + kq];     \
        }                                                                      \
        _Pragma("unroll")                                                      \
        for (int i = 0; i < 8; ++i)                                            \
            wb[i] = W[(size_t)(ks + bkb + i) * GDIM + col0 + bn];              \
    }

#define STORE_AB(buf_)                                                         \
    {   int buf = (buf_);                                                      \
        _Pragma("unroll")                                                      \
        for (int i = 0; i < 2; ++i) {                                          \
            int e = tid + i * 256;                                             \
            int rr = e >> 3, kq = (e & 7) << 2;                                \
            union { __bf16 h[4]; u32x2 q; } pa;                                \
            pa.h[0] = f2bf(va[i].x); pa.h[1] = f2bf(va[i].y);                  \
            pa.h[2] = f2bf(va[i].z); pa.h[3] = f2bf(va[i].w);                  \
            *(u32x2*)&lA[buf][rr][kq] = pa.q;                                  \
        }                                                                      \
        union { __bf16 h[8]; u32x4 q; } pb;                                    \
        _Pragma("unroll")                                                      \
        for (int i = 0; i < 8; ++i) pb.h[i] = f2bf(wb[i]);                     \
        *(u32x4*)&lB[buf][boff] = pb.q;                                        \
    }

    LOAD_AB(0);
    STORE_AB(0);
    __syncthreads();

    v8f acc0 = {}, acc1 = {};
#pragma unroll
    for (int s = 0; s < 8; ++s) {
        int cur = s & 1;
        if (s < 7) LOAD_AB((s + 1) * 32);
        if (s < 6)
            __builtin_prefetch(&X[(size_t)(row0 + (tid >> 2)) * GDIM + (s + 2) * 32], 0, 3);
        v16bf a  = frag_row40(&lA[cur][0][0], wm * 16, lane);
        v16bf b0 = frag_packed(&lB[cur][0], wn * 2, lane);
        v16bf b1 = frag_packed(&lB[cur][0], wn * 2 + 1, lane);
        acc0 = wmma_bf16(a, b0, acc0);
        acc1 = wmma_bf16(a, b1, acc1);
        if (s < 7) STORE_AB(1 - cur);
        __syncthreads();
    }
#undef LOAD_AB
#undef STORE_AB

#pragma unroll
    for (int i = 0; i < 8; ++i) {
        int m = i + ((lane >> 4) << 3);
        int r = row0 + wm * 16 + m;
        if (r < rows) {
#pragma unroll
            for (int j = 0; j < 2; ++j) {
                int c = col0 + wn * 32 + j * 16 + (lane & 15);
                float v = (j ? acc1[i] : acc0[i]) + (bias ? bias[c] : 0.f);
                if (act == 1) v = 1.f / (1.f + __expf(-v));
                Y[(size_t)r * GDIM + c] = v;
            }
        }
    }
}

// Per-graph column sums for APPNP closed-form propagation.
__global__ __launch_bounds__(256) void k_colsum(const float* __restrict__ m,
                                                float* __restrict__ S, int N) {
    int g = blockIdx.x, c = threadIdx.x;
    float acc = 0.f;
    for (int i = 0; i < N; ++i) acc += m[((size_t)g * N + i) * GDIM + c];
    S[(size_t)g * GDIM + c] = acc;
}

// h = 0.9*(S_g + m)/(N+1) + 0.1*m + x
__global__ __launch_bounds__(256) void k_combine(const float* __restrict__ m,
                                                 const float* __restrict__ S,
                                                 const float* __restrict__ x,
                                                 float* __restrict__ h, int N) {
    size_t i = (size_t)blockIdx.x * 256 + threadIdx.x;
    size_t row = i >> 8;
    int c = (int)(i & 255);
    int g = (int)(row / (size_t)N);
    float mv = m[i];
    float prop = (S[(size_t)g * GDIM + c] + mv) / (float)(N + 1);
    h[i] = 0.9f * prop + 0.1f * mv + x[i];
}

// s,t attention projections: one wave32 per row.
__global__ __launch_bounds__(256) void k_st(const float* __restrict__ xp,
                                            const float* __restrict__ aself,
                                            const float* __restrict__ aneigh,
                                            float* __restrict__ s, float* __restrict__ t,
                                            int rows) {
    int gw   = (blockIdx.x * 256 + threadIdx.x) >> 5;
    int lane = threadIdx.x & 31;
    if (gw >= rows) return;
    const float* x = xp + (size_t)gw * GDIM;
    float ss = 0.f, tt = 0.f;
#pragma unroll
    for (int i = 0; i < 8; ++i) {
        float v = x[lane + i * 32];
        ss += v * aself[lane + i * 32];
        tt += v * aneigh[lane + i * 32];
    }
#pragma unroll
    for (int o = 16; o > 0; o >>= 1) {
        ss += __shfl_xor(ss, o, 32);
        tt += __shfl_xor(tt, o, 32);
    }
    if (lane == 0) { s[gw] = ss; t[gw] = tt; }
}

// ---------------------------------------------------------------------------
// Intra-graph GAT attention (N=392): softmax(lrelu(s+t)) @ xp via WMMA.
// ---------------------------------------------------------------------------
__global__ __launch_bounds__(256) void k_attn_intra(const float* __restrict__ xp,
                                                    const float* __restrict__ h,
                                                    const float* __restrict__ s,
                                                    const float* __restrict__ t,
                                                    const float* __restrict__ gbias,
                                                    float* __restrict__ x2) {
    __shared__ __align__(16) __bf16 lE[16][40];
    __shared__ __align__(16) __bf16 lXPf[16 * 512];   // [nt][lane][16]
    __shared__ float lden[16];
    __shared__ float lsv[16];

    int gid = blockIdx.x;
    int g  = gid / 25, rt = gid % 25;
    int b  = g / NROI, r = g % NROI;
    int n0 = rt * 16;
    int vn = (392 - n0 < 16) ? (392 - n0) : 16;
    int tid = threadIdx.x, lane = tid & 31, wid = tid >> 5;

    if (tid < 16) {
        lden[tid] = 0.f;
        lsv[tid]  = (tid < vn) ? s[(size_t)g * CNODES + n0 + tid] : 0.f;
    }
    __syncthreads();

    v8f acc0 = {}, acc1 = {};

    for (int mc = 0; mc < 13; ++mc) {
#pragma unroll
        for (int j = 0; j < 2; ++j) {            // E tile 16x32 (coef numerators)
            int p = tid + j * 256;
            int n = p >> 5, mm = p & 31;
            int m = mc * 32 + mm;
            float e = 0.f;
            if (m < CNODES && n < vn) {
                float z = lsv[n] + t[(size_t)g * CNODES + m];
                z = (z > 0.f) ? z : 0.2f * z;
                e = __expf(z);
                atomicAdd(&lden[n], e);          // ds_add_f32
            }
            lE[n][mm] = f2bf(e);
        }
#pragma unroll
        for (int it = 0; it < 4; ++it) {         // XP tile -> packed frag order
            int p = tid + it * 256;
            int n = p & 255, kb = (p >> 8) << 3;
            int off = (((n >> 4) << 9) | (((n & 15) | (((kb >> 3) & 1) << 4)) << 4)) +
                      ((kb & 16) ? 8 : 0);
            union { __bf16 hh[8]; u32x4 q; } pk;
#pragma unroll
            for (int i = 0; i < 8; ++i) {
                int m = mc * 32 + kb + i;
                float v = (m < CNODES) ? xp[((size_t)g * CNODES + m) * GDIM + n] : 0.f;
                pk.hh[i] = f2bf(v);
            }
            *(u32x4*)&lXPf[off] = pk.q;
        }
        __syncthreads();
        v16bf a  = frag_row40(&lE[0][0], 0, lane);
        v16bf b0 = frag_packed(&lXPf[0], wid * 2, lane);
        v16bf b1 = frag_packed(&lXPf[0], wid * 2 + 1, lane);
        acc0 = wmma_bf16(a, b0, acc0);
        acc1 = wmma_bf16(a, b1, acc1);
        __syncthreads();
    }

#pragma unroll
    for (int i = 0; i < 8; ++i) {
        int m = i + ((lane >> 4) << 3);
        if (m < vn) {
#pragma unroll
            for (int j = 0; j < 2; ++j) {
                int c = wid * 32 + j * 16 + (lane & 15);
                float o = (j ? acc1[i] : acc0[i]) / lden[m];
                o += gbias[c];
                o = (o > 0.f) ? o : (__expf(o) - 1.f);     // elu
                size_t rowg = (size_t)g * CNODES + n0 + m;
                o += h[rowg * GDIM + c];
                size_t x2row = (size_t)b * X2NODES + (size_t)r * CNODES + n0 + m;
                x2[x2row * GDIM + c] = o;
            }
        }
    }
}

// ---------------------------------------------------------------------------
// Inter-graph GAT attention (N=13): VALU kernel. The 13x256 xp tile is staged
// memory->LDS with gfx1250 async copies (no VGPR round-trip), synchronized
// with s_wait_asynccnt + workgroup barrier.
// ---------------------------------------------------------------------------
__global__ __launch_bounds__(256) void k_attn_inter(const float* __restrict__ xp,
                                                    const float* __restrict__ h,
                                                    const float* __restrict__ s,
                                                    const float* __restrict__ t,
                                                    const float* __restrict__ gbias,
                                                    float* __restrict__ x2) {
    __shared__ __align__(16) float lxp[NROI * GDIM];
    __shared__ float le[NROI][NROI];
    __shared__ float lden[NROI];
    int g = blockIdx.x;
    int b = g / CNODES, cn = g % CNODES;
    int tid = threadIdx.x;

    // async global->LDS staging: per-lane 16B copies, tracked by ASYNCcnt.
    {
        unsigned lbase = (unsigned)(uintptr_t)&lxp[0];   // LDS aperture: low 32b = LDS offset
        const float* gsrc = xp + (size_t)g * NROI * GDIM;
        for (int i = tid; i < (NROI * GDIM) / 4; i += 256) {
            unsigned laddr = lbase + (unsigned)(i * 16);
            const float* ga = gsrc + i * 4;
            asm volatile("global_load_async_to_lds_b128 %0, %1, off"
                         :: "v"(laddr), "v"(ga) : "memory");
        }
        asm volatile("s_wait_asynccnt 0x0" ::: "memory");
    }
    if (tid < NROI * NROI) {
        int n = tid / NROI, m = tid % NROI;
        float z = s[(size_t)g * NROI + n] + t[(size_t)g * NROI + m];
        z = (z > 0.f) ? z : 0.2f * z;
        le[n][m] = __expf(z);
    }
    __syncthreads();
    if (tid < NROI) {
        float d = 0.f;
        for (int m = 0; m < NROI; ++m) d += le[tid][m];
        lden[tid] = d;
    }
    __syncthreads();

    int c = tid;
    for (int n = 0; n < NROI; ++n) {
        float acc = 0.f;
#pragma unroll
        for (int m = 0; m < NROI; ++m) acc += le[n][m] * lxp[m * GDIM + c];
        float o = acc / lden[n] + gbias[c];
        o = (o > 0.f) ? o : (__expf(o) - 1.f);
        o += h[((size_t)g * NROI + n) * GDIM + c];
        size_t x2row = (size_t)b * X2NODES + NROI * CNODES + (size_t)cn * NROI + n;
        x2[x2row * GDIM + c] = o;
    }
}

// ---------------------------------------------------------------------------
// GlobalAttentionPool: fused dual GEMM (feat & gate), sigmoid, LDS-partial
// reduction, global atomic accumulation into xf[B][512]. Double-buffered.
// ---------------------------------------------------------------------------
__global__ __launch_bounds__(256) void k_pool(const float* __restrict__ X,
                                              const float* __restrict__ Wf,
                                              const float* __restrict__ bf,
                                              const float* __restrict__ Wa,
                                              const float* __restrict__ ba,
                                              float* __restrict__ xf) {
    __shared__ __align__(16) __bf16 lA[2][64][40];
    __shared__ __align__(16) __bf16 lBf[2][2048];
    __shared__ __align__(16) __bf16 lBa[2][2048];
    __shared__ float lacc[128];
    int tid = threadIdx.x, lane = tid & 31, wid = tid >> 5;
    int wm = wid & 3, wn = wid >> 2;
    int row0 = blockIdx.x * 64, col0 = blockIdx.y * 64;
    if (tid < 128) lacc[tid] = 0.f;

    int bn  = tid & 63;
    int bkb = (tid >> 6) << 3;
    int boff = (((bn >> 4) << 9) | (((bn & 15) | (((bkb >> 3) & 1) << 4)) << 4)) +
               ((bkb & 16) ? 8 : 0);

    fx4 va[2]; float wf8[8], wa8[8];

#define P_LOAD(ks_)                                                            \
    {   int ks = (ks_);                                                        \
        _Pragma("unroll")                                                      \
        for (int i = 0; i < 2; ++i) {                                          \
            int e = tid + i * 256;                                             \
            int rr = e >> 3, kq = (e & 7) << 2;                                \
            va[i] = *(const fx4*)&X[(size_t)(row0 + rr) * GDIM + ks + kq];     \
        }                                                                      \
        _Pragma("unroll")                                                      \
        for (int i = 0; i < 8; ++i) {                                          \
            wf8[i] = Wf[(size_t)(ks + bkb + i) * 512 + col0 + bn];             \
            wa8[i] = Wa[(size_t)(ks + bkb + i) * 512 + col0 + bn];             \
        }                                                                      \
    }

#define P_STORE(buf_)                                                          \
    {   int buf = (buf_);                                                      \
        _Pragma("unroll")                                                      \
        for (int i = 0; i < 2; ++i) {                                          \
            int e = tid + i * 256;                                             \
            int rr = e >> 3, kq = (e & 7) << 2;                                \
            union { __bf16 h[4]; u32x2 q; } pa;                                \
            pa.h[0] = f2bf(va[i].x); pa.h[1] = f2bf(va[i].y);                  \
            pa.h[2] = f2bf(va[i].z); pa.h[3] = f2bf(va[i].w);                  \
            *(u32x2*)&lA[buf][rr][kq] = pa.q;                                  \
        }                                                                      \
        union { __bf16 h[8]; u32x4 q; } pf, pg;                                \
        _Pragma("unroll")                                                      \
        for (int i = 0; i < 8; ++i) { pf.h[i] = f2bf(wf8[i]); pg.h[i] = f2bf(wa8[i]); } \
        *(u32x4*)&lBf[buf][boff] = pf.q;                                       \
        *(u32x4*)&lBa[buf][boff] = pg.q;                                       \
    }

    P_LOAD(0);
    P_STORE(0);
    __syncthreads();

    v8f af0 = {}, af1 = {}, ag0 = {}, ag1 = {};
#pragma unroll
    for (int s = 0; s < 8; ++s) {
        int cur = s & 1;
        if (s < 7) P_LOAD((s + 1) * 32);
        if (s < 6)
            __builtin_prefetch(&X[(size_t)(row0 + (tid >> 2)) * GDIM + (s + 2) * 32], 0, 3);
        v16bf a  = frag_row40(&lA[cur][0][0], wm * 16, lane);
        v16bf b0 = frag_packed(&lBf[cur][0], wn * 2, lane);
        v16bf b1 = frag_packed(&lBf[cur][0], wn * 2 + 1, lane);
        v16bf c0 = frag_packed(&lBa[cur][0], wn * 2, lane);
        v16bf c1 = frag_packed(&lBa[cur][0], wn * 2 + 1, lane);
        af0 = wmma_bf16(a, b0, af0);
        af1 = wmma_bf16(a, b1, af1);
        ag0 = wmma_bf16(a, c0, ag0);
        ag1 = wmma_bf16(a, c1, ag1);
        if (s < 7) P_STORE(1 - cur);
        __syncthreads();
    }
#undef P_LOAD
#undef P_STORE

    int b0r = row0 / X2NODES;
#pragma unroll
    for (int i = 0; i < 8; ++i) {
        int m = i + ((lane >> 4) << 3);
        int rrow = row0 + wm * 16 + m;
#pragma unroll
        for (int j = 0; j < 2; ++j) {
            int c = col0 + wn * 32 + j * 16 + (lane & 15);
            float fv = (j ? af1[i] : af0[i]) + bf[c];
            float gv = (j ? ag1[i] : ag0[i]) + ba[c];
            gv = 1.f / (1.f + __expf(-gv));
            int bb = rrow / X2NODES - b0r;
            atomicAdd(&lacc[bb * 64 + (c - col0)], fv * gv);
        }
    }
    __syncthreads();
    if (tid < 128) {
        int bb = tid >> 6, cc = tid & 63;
        int bidx = b0r + bb;
        float v = lacc[tid];
        if (bidx < BATCH && v != 0.f)
            atomicAdd(&xf[(size_t)bidx * 512 + col0 + cc], v);
    }
}

__global__ void k_zero(float* __restrict__ p, int n) {
    int i = blockIdx.x * 256 + threadIdx.x;
    if (i < n) p[i] = 0.f;
}

// BN (inference) + dense(512x200) + softmax. One block per batch row.
__global__ __launch_bounds__(256) void k_head(const float* __restrict__ xf,
                                              const float* __restrict__ gamma,
                                              const float* __restrict__ beta,
                                              const float* __restrict__ mean,
                                              const float* __restrict__ var,
                                              const float* __restrict__ dw,
                                              const float* __restrict__ db,
                                              float* __restrict__ out) {
    __shared__ float xn[512];
    __shared__ float lg[200];
    __shared__ float red[2];
    int b = blockIdx.x, tid = threadIdx.x;
    for (int i = tid; i < 512; i += 256)
        xn[i] = (xf[(size_t)b * 512 + i] - mean[i]) * rsqrtf(var[i] + 1e-3f) * gamma[i] + beta[i];
    __syncthreads();
    if (tid < 200) {
        float acc = db[tid];
        for (int k = 0; k < 512; ++k) acc += xn[k] * dw[(size_t)k * 200 + tid];
        lg[tid] = acc;
    }
    __syncthreads();
    if (tid == 0) {
        float mx = -1e30f;
        for (int i = 0; i < 200; ++i) mx = fmaxf(mx, lg[i]);
        float su = 0.f;
        for (int i = 0; i < 200; ++i) su += __expf(lg[i] - mx);
        red[0] = mx; red[1] = su;
    }
    __syncthreads();
    if (tid < 200)
        out[(size_t)b * 200 + tid] = __expf(lg[tid] - red[0]) / red[1];
}

// ---------------------------------------------------------------------------
extern "C" void kernel_launch(void* const* d_in, const int* in_sizes, int n_in,
                              void* d_out, int out_size, void* d_ws, size_t ws_size,
                              hipStream_t stream) {
    (void)in_sizes; (void)n_in; (void)out_size; (void)ws_size;
    const float* base    = (const float*)d_in[0];
    const float* appnp_w = (const float*)d_in[1];
    const float* appnp_b = (const float*)d_in[2];
    const float* gat_k   = (const float*)d_in[3];
    const float* gat_as  = (const float*)d_in[4];
    const float* gat_an  = (const float*)d_in[5];
    const float* gat_b   = (const float*)d_in[6];
    const float* pf_w    = (const float*)d_in[7];
    const float* pf_b    = (const float*)d_in[8];
    const float* pa_w    = (const float*)d_in[9];
    const float* pa_b    = (const float*)d_in[10];
    const float* bn_g    = (const float*)d_in[11];
    const float* bn_be   = (const float*)d_in[12];
    const float* bn_m    = (const float*)d_in[13];
    const float* bn_v    = (const float*)d_in[14];
    const float* dw      = (const float*)d_in[15];
    const float* db      = (const float*)d_in[16];
    float* out = (float*)d_out;

    float* ws = (float*)d_ws;
    const size_t NI = (size_t)ROWS_G * GDIM;
    float* nintra = ws;
    float* ninter = nintra + NI;
    float* bm     = ninter + NI;
    float* bh     = bm + NI;
    float* bxp    = bh + NI;
    float* bs     = bxp + NI;
    float* bt     = bs + ROWS_G;
    float* S      = bt + ROWS_G;
    float* x2     = S + (size_t)3136 * GDIM;
    float* xf     = x2 + (size_t)POOLROWS * GDIM;

    dim3 gemm_grid(ROWS_G / 64, 4);

    k_roi_nodes<<<BATCH * NROI * 49, 256, 0, stream>>>(base, nintra, ninter);

    // intra-ROI graphs (104 graphs x 392 nodes)
    k_gemm256<<<gemm_grid, 256, 0, stream>>>(nintra, appnp_w, appnp_b, bm, ROWS_G, 1);
    k_colsum<<<BATCH * NROI, 256, 0, stream>>>(bm, S, CNODES);
    k_combine<<<ROWS_G, 256, 0, stream>>>(bm, S, nintra, bh, CNODES);
    k_gemm256<<<gemm_grid, 256, 0, stream>>>(bh, gat_k, nullptr, bxp, ROWS_G, 0);
    k_st<<<ROWS_G / 8, 256, 0, stream>>>(bxp, gat_as, gat_an, bs, bt, ROWS_G);
    k_attn_intra<<<BATCH * NROI * 25, 256, 0, stream>>>(bxp, bh, bs, bt, gat_b, x2);

    // inter-ROI graphs (3136 graphs x 13 nodes)
    k_gemm256<<<gemm_grid, 256, 0, stream>>>(ninter, appnp_w, appnp_b, bm, ROWS_G, 1);
    k_colsum<<<BATCH * CNODES, 256, 0, stream>>>(bm, S, NROI);
    k_combine<<<ROWS_G, 256, 0, stream>>>(bm, S, ninter, bh, NROI);
    k_gemm256<<<gemm_grid, 256, 0, stream>>>(bh, gat_k, nullptr, bxp, ROWS_G, 0);
    k_st<<<ROWS_G / 8, 256, 0, stream>>>(bxp, gat_as, gat_an, bs, bt, ROWS_G);
    k_attn_inter<<<BATCH * CNODES, 256, 0, stream>>>(bxp, bh, bs, bt, gat_b, x2);

    // global attention pool + head
    k_zero<<<16, 256, 0, stream>>>(xf, BATCH * 512);
    k_pool<<<dim3(POOLROWS / 64, 8), 256, 0, stream>>>(x2, pf_w, pf_b, pa_w, pa_b, xf);
    k_head<<<BATCH, 256, 0, stream>>>(xf, bn_g, bn_be, bn_m, bn_v, dw, db, out);
}
